// LSTMDecoder_89661737271961
// MI455X (gfx1250) — matrix operand order repaired and checked by reading, not compiled, not run on hardware
//
#include <hip/hip_runtime.h>
#include <hip/hip_bf16.h>
#include <math.h>

#define B_  32
#define H_  1024
#define V_  32000
#define T_  64
#define G4H (4 * H_)

typedef __attribute__((ext_vector_type(2)))  float        v2f;
typedef __attribute__((ext_vector_type(8)))  float        v8f;
typedef __attribute__((ext_vector_type(8)))  __bf16       v8bf;
typedef __attribute__((ext_vector_type(16))) __bf16       v16bf;
typedef __attribute__((ext_vector_type(4)))  unsigned int v4u;
typedef __attribute__((ext_vector_type(8)))  int          v8i_;
typedef __attribute__((ext_vector_type(4)))  int          v4i_;

// ---------------- workspace layout (bytes) ----------------
static constexpr size_t SZ_WOUT_BF = (size_t)V_ * H_ * 2;          // 65,536,000
static constexpr size_t OFF_WOUT   = 0;
static constexpr size_t OFF_H      = OFF_WOUT + SZ_WOUT_BF;        // fp32 h [32][1024]
static constexpr size_t OFF_C      = OFF_H    + (size_t)B_ * H_ * 4;
static constexpr size_t OFF_HBF    = OFF_C    + (size_t)B_ * H_ * 4;
static constexpr size_t OFF_GATES  = OFF_HBF  + (size_t)B_ * H_ * 2;
static constexpr size_t OFF_STATS  = OFF_GATES + (size_t)B_ * G4H * 4;

// ---------------- TDM: async 2D bf16 tile load, global -> LDS ----------------
// D# per CDNA5 ISA: group0 {count=1, lds_addr, global_addr,
// type=2}; group1 {data_size=1(2B), tensor_dim0/1, tile_dim0/1, dim0_stride};
// groups 2/3 zero (2D tensor). Issued once per wave; EXEC ignored by TDM.
// Toolchain uses the 6-arg builtin form (clang-23 / therock-10.0).
__device__ __forceinline__ void tdm_load_2d_bf16(const void* gptr, unsigned lds_off,
                                                 unsigned dim0, unsigned dim1,
                                                 unsigned stride0) {
    unsigned long long ga = (unsigned long long)(uintptr_t)gptr;
    v4u g0;
    g0[0] = 1u;                                             // count=1
    g0[1] = lds_off;                                        // lds_addr (bytes)
    g0[2] = (unsigned)(ga & 0xFFFFFFFFu);                   // global_addr[31:0]
    g0[3] = (unsigned)((ga >> 32) & 0x01FFFFFFu) | (2u << 30); // addr[56:32] | type=2
    v8i_ g1;
    g1[0] = (int)(1u << 16);                                // data_size = 1 -> 2 bytes
    g1[1] = (int)((dim0 & 0xFFFFu) << 16);                  // tensor_dim0[15:0]
    g1[2] = (int)(((dim0 >> 16) & 0xFFFFu) | ((dim1 & 0xFFFFu) << 16)); // dim0 hi | dim1 lo
    g1[3] = (int)(((dim1 >> 16) & 0xFFFFu) | ((dim0 & 0xFFFFu) << 16)); // dim1 hi | tile_dim0
    g1[4] = (int)(dim1 & 0xFFFFu);                          // tile_dim1 | tile_dim2=0
    g1[5] = (int)stride0;                                   // tensor_dim0_stride[31:0]
    g1[6] = 0;
    g1[7] = 0;
    v4i_ z4 = {0, 0, 0, 0};
    v8i_ z8 = {0, 0, 0, 0, 0, 0, 0, 0};
    __builtin_amdgcn_tensor_load_to_lds(g0, g1, z4, z4, z8, 0);
}

// ---------------- one-time prep ----------------
__global__ void __launch_bounds__(256) k_cvt_wout(const float* __restrict__ W,
                                                  __bf16* __restrict__ Wb) {
    size_t i = ((size_t)blockIdx.x * blockDim.x + threadIdx.x) * 4;
    if (i + 3 < (size_t)V_ * H_) {
        float4 f = *(const float4*)(W + i);
        union { __bf16 b[4]; unsigned long long u; } u;
        u.b[0] = (__bf16)f.x; u.b[1] = (__bf16)f.y;
        u.b[2] = (__bf16)f.z; u.b[3] = (__bf16)f.w;
        *(unsigned long long*)(Wb + i) = u.u;
    }
}

__global__ void __launch_bounds__(256) k_init_state(const float* __restrict__ h0,
                                                    const float* __restrict__ c0,
                                                    float* __restrict__ h,
                                                    float* __restrict__ c,
                                                    __bf16* __restrict__ hbf) {
    int i = blockIdx.x * blockDim.x + threadIdx.x;   // B_*H_ = 32768 threads
    float hv = h0[i];
    h[i] = hv;
    c[i] = c0[i];
    hbf[i] = (__bf16)hv;
}

// ---------------- recurrent gates GEMM: fp32 WMMA 16x16x4 ----------------
// gates[32][4096] = h[32][1024] @ W_hh^T  (bias added in k_cell)
// grid = (32,1) x 128 threads; wave covers full M=32 x N=32 so W_hh is
// streamed from L2 exactly once per step.
__global__ void __launch_bounds__(128) k_gates(const float* __restrict__ h,
                                               const float* __restrict__ Whh,
                                               float* __restrict__ gates) {
    const int lane = threadIdx.x & 31;
    const int wave = threadIdx.x >> 5;
    const int n0   = blockIdx.x * 128 + wave * 32;
    const int kb   = (lane >= 16) ? 2 : 0;          // half-wave K offset (f32 frags)

    const float* ha0 = h   + (size_t)(lane & 15) * H_ + kb;         // rows 0..15
    const float* ha1 = ha0 + 16 * H_;                               // rows 16..31
    const float* bp  = Whh + (size_t)(n0 + (lane & 15)) * H_ + kb;  // B col = Whh row

    v8f acc00 = {}, acc01 = {}, acc10 = {}, acc11 = {};
    for (int k0 = 0; k0 < H_; k0 += 4) {
        v2f a0 = *(const v2f*)(ha0 + k0);
        v2f a1 = *(const v2f*)(ha1 + k0);
        v2f b0 = *(const v2f*)(bp + k0);
        v2f b1 = *(const v2f*)(bp + 16 * H_ + k0);
        acc00 = __builtin_amdgcn_wmma_f32_16x16x4_f32(false, a0, false, b0, (short)0, acc00, false, false);
        acc01 = __builtin_amdgcn_wmma_f32_16x16x4_f32(false, a0, false, b1, (short)0, acc01, false, false);
        acc10 = __builtin_amdgcn_wmma_f32_16x16x4_f32(false, a1, false, b0, (short)0, acc10, false, false);
        acc11 = __builtin_amdgcn_wmma_f32_16x16x4_f32(false, a1, false, b1, (short)0, acc11, false, false);
    }
    // D layout: VGPR r -> M = 16*mi + r + (lane>=16 ? 8 : 0), N = lane & 15
    const int mAdd = (lane >= 16) ? 8 : 0;
    const int ncol = lane & 15;
    #pragma unroll
    for (int r = 0; r < 8; ++r) {
        float* g0 = gates + (size_t)(r + mAdd) * G4H + n0 + ncol;
        float* g1 = gates + (size_t)(16 + r + mAdd) * G4H + n0 + ncol;
        g0[0]  = acc00[r];
        g0[16] = acc01[r];
        g1[0]  = acc10[r];
        g1[16] = acc11[r];
    }
}

// ---------------- LSTM cell pointwise update ----------------
__global__ void __launch_bounds__(256) k_cell(const float* __restrict__ gates,
                                              const float* __restrict__ bih,
                                              const float* __restrict__ bhh,
                                              float* __restrict__ h,
                                              float* __restrict__ c,
                                              __bf16* __restrict__ hbf) {
    int idx = blockIdx.x * blockDim.x + threadIdx.x;   // B_*H_ threads
    int b = idx >> 10, k = idx & (H_ - 1);
    const float* g = gates + (size_t)b * G4H;
    float gi = g[k]          + bih[k]          + bhh[k];
    float gf = g[H_ + k]     + bih[H_ + k]     + bhh[H_ + k];
    float gg = g[2 * H_ + k] + bih[2 * H_ + k] + bhh[2 * H_ + k];
    float go = g[3 * H_ + k] + bih[3 * H_ + k] + bhh[3 * H_ + k];
    float si = 1.0f / (1.0f + __expf(-gi));
    float sf = 1.0f / (1.0f + __expf(-gf));
    float so = 1.0f / (1.0f + __expf(-go));
    float cn = sf * c[idx] + si * tanhf(gg);
    float hn = so * tanhf(cn);
    c[idx] = cn;
    h[idx] = hn;
    hbf[idx] = (__bf16)hn;
}

// ---------------- output projection GEMM: bf16 WMMA 16x16x32 ----------------
// logits[32][32000] = h_bf16 @ W_out_bf16^T + b_out, raw to d_out[b][t][:]
// grid = (250,1) x 128 threads; wave covers full M=32 x N=32 -> W_out streamed
// from L2 exactly once per step. A (h_bf16, 64 KB) staged to LDS via TDM.
__global__ void __launch_bounds__(128) k_proj(const __bf16* __restrict__ hbf,
                                              const __bf16* __restrict__ Wb,
                                              const float* __restrict__ bout,
                                              float* __restrict__ out, int t) {
    __shared__ __bf16 smem[B_ * H_];                 // 64 KB A tile
    const int lane = threadIdx.x & 31;
    const int wave = threadIdx.x >> 5;
    const int n0   = blockIdx.x * 128 + wave * 32;
    const int kbA  = (lane >= 16) ? 16 : 0;          // A frag half-wave K offset
    const int kbB  = (lane >= 16) ? 16 : 0;          // B frag half-wave K offset

    if (wave == 0) {
        tdm_load_2d_bf16(hbf, (unsigned)(uintptr_t)smem, H_, B_, H_);
        __builtin_amdgcn_s_wait_tensorcnt(0);
    }
    __syncthreads();

    const __bf16* ha0 = smem + (size_t)(lane & 15) * H_ + kbA;      // rows 0..15
    const __bf16* ha1 = ha0 + 16 * H_;                              // rows 16..31
    const __bf16* bp  = Wb + (size_t)(n0 + (lane & 15)) * H_ + kbB; // W_out row

    v8f acc00 = {}, acc01 = {}, acc10 = {}, acc11 = {};
    for (int k0 = 0; k0 < H_; k0 += 32) {
        __builtin_prefetch((const void*)(bp + k0 + 512), 0, 3);     // stream W_out ahead
        union { v16bf v; v8bf p[2]; } a0, a1;
        a0.p[0] = *(const v8bf*)(ha0 + k0);
        a0.p[1] = *(const v8bf*)(ha0 + k0 + 8);
        a1.p[0] = *(const v8bf*)(ha1 + k0);
        a1.p[1] = *(const v8bf*)(ha1 + k0 + 8);
        v16bf b0 = *(const v16bf*)(bp + k0);
        v16bf b1 = *(const v16bf*)(bp + 16 * H_ + k0);
        acc00 = __builtin_amdgcn_wmma_f32_16x16x32_bf16(false, a0.v, false, b0, (short)0, acc00, false, false);
        acc01 = __builtin_amdgcn_wmma_f32_16x16x32_bf16(false, a0.v, false, b1, (short)0, acc01, false, false);
        acc10 = __builtin_amdgcn_wmma_f32_16x16x32_bf16(false, a1.v, false, b0, (short)0, acc10, false, false);
        acc11 = __builtin_amdgcn_wmma_f32_16x16x32_bf16(false, a1.v, false, b1, (short)0, acc11, false, false);
    }
    const int mAdd = (lane >= 16) ? 8 : 0;
    const int ncol = lane & 15;
    float bo0 = bout[n0 + ncol];
    float bo1 = bout[n0 + 16 + ncol];
    #pragma unroll
    for (int r = 0; r < 8; ++r) {
        float* o0 = out + ((size_t)(r + mAdd) * T_ + t) * V_ + n0 + ncol;
        float* o1 = out + ((size_t)(16 + r + mAdd) * T_ + t) * V_ + n0 + ncol;
        o0[0]  = acc00[r] + bo0;
        o0[16] = acc01[r] + bo1;
        o1[0]  = acc10[r] + bo0;
        o1[16] = acc11[r] + bo1;
    }
}

// ---------------- row-wise max + logsumexp (one block per batch row) ----------------
__global__ void __launch_bounds__(256) k_rowred(const float* __restrict__ out,
                                                float* __restrict__ stats, int t) {
    __shared__ float s[256];
    const int b = blockIdx.x, tid = threadIdx.x;
    const float* z = out + ((size_t)b * T_ + t) * V_;
    float m = -INFINITY;
    for (int v = tid; v < V_; v += 256) m = fmaxf(m, z[v]);
    s[tid] = m; __syncthreads();
    for (int off = 128; off > 0; off >>= 1) {
        if (tid < off) s[tid] = fmaxf(s[tid], s[tid + off]);
        __syncthreads();
    }
    m = s[0]; __syncthreads();
    float sum = 0.0f;
    for (int v = tid; v < V_; v += 256) sum += __expf(z[v] - m);
    s[tid] = sum; __syncthreads();
    for (int off = 128; off > 0; off >>= 1) {
        if (tid < off) s[tid] += s[tid + off];
        __syncthreads();
    }
    if (tid == 0) stats[b] = m + __logf(s[0]);
}

// ---------------- normalize: logits -> log-probs ----------------
__global__ void __launch_bounds__(256) k_norm(float* __restrict__ out,
                                              const float* __restrict__ stats, int t) {
    int i = blockIdx.x * blockDim.x + threadIdx.x;   // B_*V_ threads
    if (i < B_ * V_) {
        int b = i / V_, v = i - b * V_;
        out[((size_t)b * T_ + t) * V_ + v] -= stats[b];
    }
}

extern "C" void kernel_launch(void* const* d_in, const int* in_sizes, int n_in,
                              void* d_out, int out_size, void* d_ws, size_t ws_size,
                              hipStream_t stream) {
    const float* h0   = (const float*)d_in[0];
    const float* c0   = (const float*)d_in[1];
    // d_in[2] = W_ih: unused — decoder input is all-zeros, contribution is b_ih.
    const float* Whh  = (const float*)d_in[3];
    const float* bih  = (const float*)d_in[4];
    const float* bhh  = (const float*)d_in[5];
    const float* Wout = (const float*)d_in[6];
    const float* bout = (const float*)d_in[7];
    float* out = (float*)d_out;

    char* ws = (char*)d_ws;
    __bf16* Wb    = (__bf16*)(ws + OFF_WOUT);
    float*  h     = (float*)(ws + OFF_H);
    float*  c     = (float*)(ws + OFF_C);
    __bf16* hbf   = (__bf16*)(ws + OFF_HBF);
    float*  gates = (float*)(ws + OFF_GATES);
    float*  stats = (float*)(ws + OFF_STATS);

    k_cvt_wout<<<(V_ * H_) / (256 * 4), 256, 0, stream>>>(Wout, Wb);
    k_init_state<<<(B_ * H_) / 256, 256, 0, stream>>>(h0, c0, h, c, hbf);

    for (int t = 0; t < T_; ++t) {
        k_gates<<<dim3(32, 1), 128, 0, stream>>>(h, Whh, gates);
        k_cell<<<(B_ * H_) / 256, 256, 0, stream>>>(gates, bih, bhh, h, c, hbf);
        k_proj<<<dim3(250, 1), 128, 0, stream>>>(hbf, Wb, bout, out, t);
        k_rowred<<<B_, 256, 0, stream>>>(out, stats, t);
        k_norm<<<(B_ * V_ + 255) / 256, 256, 0, stream>>>(out, stats, t);
    }
}
